// PointNet_14018773254577
// MI455X (gfx1250) — matrix operand order repaired and checked by reading, not compiled, not run on hardware
//
#include <hip/hip_runtime.h>

typedef __attribute__((ext_vector_type(16))) __bf16 v16bf;
typedef __attribute__((ext_vector_type(8)))  float  v8f;
typedef __attribute__((ext_vector_type(4)))  float  vf4;

#define NCH 32

// ---------------------------------------------------------------------------
// Per-node precompute for conv1 (h == pos, 6->32 first layer folded per node):
//   U[n][c] = sum_k pos[n][k]*(W1a[k][c] + W1a[3+k][c]) + b1a[c]
//   V[n][c] = sum_k pos[n][k]* W1a[3+k][c]
// so per-edge layer-1 preact = U[src] - V[dst].
// ---------------------------------------------------------------------------
__global__ __launch_bounds__(256) void pre1_kernel(
    const float* __restrict__ pos, const float* __restrict__ w1a,
    const float* __restrict__ b1a, float* __restrict__ U, float* __restrict__ V,
    int nNodes) {
  int idx = blockIdx.x * blockDim.x + threadIdx.x;
  if (idx >= nNodes * NCH) return;
  int nd = idx >> 5, c = idx & 31;
  float p0 = pos[nd * 3 + 0], p1 = pos[nd * 3 + 1], p2 = pos[nd * 3 + 2];
  float wh0 = w1a[0 * 32 + c], wh1 = w1a[1 * 32 + c], wh2 = w1a[2 * 32 + c];
  float wp0 = w1a[3 * 32 + c], wp1 = w1a[4 * 32 + c], wp2 = w1a[5 * 32 + c];
  float v = p0 * wp0 + p1 * wp1 + p2 * wp2;
  float u = p0 * (wh0 + wp0) + p1 * (wh1 + wp1) + p2 * (wh2 + wp2) + b1a[c];
  U[idx] = u;
  V[idx] = v;
}

// ---------------------------------------------------------------------------
// Per-node precompute for conv2 (35->32 first layer, h from conv1):
//   U[n][c] = h[n]@W1b[0:32][c] + pos[n]@W1b[32:35][c] + b1b[c]
//   V[n][c] = pos[n]@W1b[32:35][c]
// ---------------------------------------------------------------------------
__global__ __launch_bounds__(256) void pre2_kernel(
    const float* __restrict__ pos, const float* __restrict__ H1,
    const float* __restrict__ w1b, const float* __restrict__ b1b,
    float* __restrict__ U, float* __restrict__ V, int nNodes) {
  int idx = blockIdx.x * blockDim.x + threadIdx.x;
  if (idx >= nNodes * NCH) return;
  int nd = idx >> 5, c = idx & 31;
  float acc = b1b[c];
  const float* h = H1 + (unsigned)nd * NCH;
#pragma unroll 8
  for (int k = 0; k < 32; ++k) acc += h[k] * w1b[k * 32 + c];
  float p0 = pos[nd * 3 + 0], p1 = pos[nd * 3 + 1], p2 = pos[nd * 3 + 2];
  float wp0 = w1b[32 * 32 + c], wp1 = w1b[33 * 32 + c], wp2 = w1b[34 * 32 + c];
  float v = p0 * wp0 + p1 * wp1 + p2 * wp2;
  U[idx] = acc + v;
  V[idx] = v;
}

// Convert the two 32x32 second-layer weight matrices to bf16 (row-major [K][N]).
__global__ void wconv_kernel(const float* __restrict__ w2a,
                             const float* __restrict__ w2b,
                             __bf16* __restrict__ wa, __bf16* __restrict__ wb) {
  int i = blockIdx.x * blockDim.x + threadIdx.x;
  if (i < 1024)
    wa[i] = (__bf16)w2a[i];
  else if (i < 2048)
    wb[i - 1024] = (__bf16)w2b[i - 1024];
}

__global__ void zero_kernel(float* __restrict__ p, long n) {
  long i = (long)blockIdx.x * blockDim.x + threadIdx.x;
  long stride = (long)gridDim.x * blockDim.x;
  for (; i < n; i += stride) p[i] = 0.0f;
}

// ---------------------------------------------------------------------------
// Per-edge second MLP layer + scatter-max, fully on bf16 WMMA.
// One wave32 handles a tile of 16 edges:
//   A[16x32] = relu(U[src] - V[dst]) (bf16)
//   C       = bias broadcast (b2[n] per lane, valid for all D rows)
//   D = A@B + C  -> 2x V_WMMA_F32_16X16X32_BF16 (bias folded into C)
//   D -> atomicMax into H[dst][c]  (int-bit max; H zero-initialized, which
//        also implements where(isfinite)->0 and the later ReLU)
// Hot loop covers FULL tiles only (guard-free body, no exec juggling);
// a cold scalar path run by one wave covers the <16 remainder edges.
// ---------------------------------------------------------------------------
__global__ __launch_bounds__(256) void edge_conv_kernel(
    const float* __restrict__ U, const float* __restrict__ V,
    const __bf16* __restrict__ W2,  // [32][32] row-major K x N, bf16
    const float* __restrict__ b2,   // [32]
    const int* __restrict__ esrc, const int* __restrict__ edst,
    float* __restrict__ H,          // [N][32], pre-zeroed, atomicMax target
    int nEdges) {
  const int lane = threadIdx.x & 31;
  const int waveInBlock = threadIdx.x >> 5;
  const int wavesPerBlock = blockDim.x >> 5;
  const int globalWave = blockIdx.x * wavesPerBlock + waveInBlock;
  const int totalWaves = gridDim.x * wavesPerBlock;
  const int n = lane & 15;   // matrix row (A) / column (B, D) owned by lane
  const int kh = lane >> 4;  // lane half selects K sub-range

  // B-matrix register images (same K order as A: lane half kh holds K in
  // {kh*8..kh*8+7} and {16+kh*8..16+kh*8+7}, N = lane&15). Loaded once, reused for all tiles.
  v16bf blo, bhi;
#pragma unroll
  for (int i = 0; i < 16; ++i) {
    int k = (i < 8) ? (kh * 8 + i) : (16 + kh * 8 + (i - 8));
    blo[i] = W2[k * 32 + n];
    bhi[i] = W2[k * 32 + 16 + n];
  }
  // Bias folded into the WMMA C operand: C/D layout has column n fixed per
  // lane and rows across elements, so a per-lane broadcast of b2[n] is the
  // correct bias matrix.
  const float biasLo = b2[n];
  const float biasHi = b2[16 + n];
  v8f cLo, cHi;
#pragma unroll
  for (int r = 0; r < 8; ++r) { cLo[r] = biasLo; cHi[r] = biasHi; }

  int* __restrict__ Hi = (int*)H;
  const int nFull = nEdges >> 4;  // full 16-edge tiles
  const int tileStride = totalWaves;

  for (int t = globalWave; t < nFull; t += tileStride) {
    const int tb = t << 4;
    const int e = tb + n;
    const int s = esrc[e];
    const int d = edst[e];

    // Prefetch the next tile's index cachelines this wave will consume.
    {
      int pn = tb + tileStride * 16;
      if (pn >= nEdges) pn = nEdges - 1;
      __builtin_prefetch(esrc + pn, 0, 0);
      __builtin_prefetch(edst + pn, 0, 0);
    }

    // A-matrix 16-bit layout: lane half kh holds K in {kh*8..kh*8+7} and
    // {16+kh*8..16+kh*8+7}; element order matches VGPR0..7 packing.
    const unsigned ub = (unsigned)s * NCH;
    const unsigned vb = (unsigned)d * NCH;
    const unsigned k0 = (unsigned)(kh * 8);
    const unsigned k1 = (unsigned)(16 + kh * 8);
    vf4 u0 = *(const vf4*)(U + ub + k0), u1 = *(const vf4*)(U + ub + k0 + 4);
    vf4 u2 = *(const vf4*)(U + ub + k1), u3 = *(const vf4*)(U + ub + k1 + 4);
    vf4 v0 = *(const vf4*)(V + vb + k0), v1 = *(const vf4*)(V + vb + k0 + 4);
    vf4 v2 = *(const vf4*)(V + vb + k1), v3 = *(const vf4*)(V + vb + k1 + 4);

    v16bf a;
#pragma unroll
    for (int j = 0; j < 4; ++j) {
      a[j]      = (__bf16)fmaxf(u0[j] - v0[j], 0.0f);
      a[4 + j]  = (__bf16)fmaxf(u1[j] - v1[j], 0.0f);
      a[8 + j]  = (__bf16)fmaxf(u2[j] - v2[j], 0.0f);
      a[12 + j] = (__bf16)fmaxf(u3[j] - v3[j], 0.0f);
    }

    v8f dlo = __builtin_amdgcn_wmma_f32_16x16x32_bf16(
        false, a, false, blo, (short)0, cLo, false, false);
    v8f dhi = __builtin_amdgcn_wmma_f32_16x16x32_bf16(
        false, a, false, bhi, (short)0, cHi, false, false);

    // Scatter: D element r belongs to edge row m2 = r + 8*kh, column n.
    // That edge's dst lives in lane m2's `d` register. Batch all shuffles
    // first so the ds_bpermutes pipeline, then fire the atomics.
    int dsts[8];
#pragma unroll
    for (int r = 0; r < 8; ++r) dsts[r] = __shfl(d, r + 8 * kh, 32);
#pragma unroll
    for (int r = 0; r < 8; ++r) {
      unsigned off = (unsigned)dsts[r] * NCH;
      atomicMax(Hi + off + n, __float_as_int(dlo[r]));
      atomicMax(Hi + off + 16 + n, __float_as_int(dhi[r]));
    }
  }

  // Cold tail: remainder edges (< 16), scalar f32 path, one wave only.
  const int tailBase = nFull << 4;
  const int tail = nEdges - tailBase;
  if (tail > 0 && globalWave == 0 && lane < tail) {
    const int e = tailBase + lane;
    const int s = esrc[e];
    const int d = edst[e];
    float tv[NCH];
#pragma unroll 8
    for (int k = 0; k < NCH; ++k)
      tv[k] = fmaxf(U[(unsigned)s * NCH + k] - V[(unsigned)d * NCH + k], 0.0f);
    for (int c = 0; c < NCH; ++c) {
      float acc = b2[c];
#pragma unroll 8
      for (int k = 0; k < NCH; ++k) acc += tv[k] * (float)W2[k * 32 + c];
      atomicMax(Hi + (unsigned)d * NCH + c, __float_as_int(acc));
    }
  }
}

// ---------------------------------------------------------------------------
// Global max pool: LDS-staged per-block [64][32] max, then global atomicMax.
// Values are >= 0 (post-ReLU), so int-bit max on 0-initialized memory is exact.
// ---------------------------------------------------------------------------
__global__ __launch_bounds__(256) void pool_kernel(
    const float* __restrict__ H2, const int* __restrict__ batch,
    float* __restrict__ G, int nNodes) {
  __shared__ int smax[64 * 32];
  for (int i = threadIdx.x; i < 64 * 32; i += blockDim.x) smax[i] = 0;
  __syncthreads();
  const int c = threadIdx.x & 31;
  const int row = threadIdx.x >> 5;
  const int rowsPerBlock = blockDim.x >> 5;
  for (int nd = blockIdx.x * rowsPerBlock + row; nd < nNodes;
       nd += gridDim.x * rowsPerBlock) {
    float h = H2[(unsigned)nd * NCH + c];
    int b = batch[nd];
    atomicMax(&smax[b * 32 + c], __float_as_int(h));
  }
  __syncthreads();
  for (int i = threadIdx.x; i < 64 * 32; i += blockDim.x)
    atomicMax((int*)&G[i], smax[i]);
}

// Final classifier: out[g][c] = G[g] @ wc[:,c] + bc[c]   (64 x 40)
__global__ void cls_kernel(const float* __restrict__ G,
                           const float* __restrict__ wc,
                           const float* __restrict__ bc,
                           float* __restrict__ out) {
  int o = blockIdx.x * blockDim.x + threadIdx.x;
  if (o >= 64 * 40) return;
  int g = o / 40, c = o - g * 40;
  float acc = bc[c];
#pragma unroll 8
  for (int k = 0; k < 32; ++k) acc += G[g * 32 + k] * wc[k * 40 + c];
  out[o] = acc;
}

extern "C" void kernel_launch(void* const* d_in, const int* in_sizes, int n_in,
                              void* d_out, int out_size, void* d_ws,
                              size_t ws_size, hipStream_t stream) {
  const float* pos   = (const float*)d_in[0];
  const int*   eidx  = (const int*)d_in[1];
  const int*   batch = (const int*)d_in[2];
  const float* w1a = (const float*)d_in[3];
  const float* b1a = (const float*)d_in[4];
  const float* w2a = (const float*)d_in[5];
  const float* b2a = (const float*)d_in[6];
  const float* w1b = (const float*)d_in[7];
  const float* b1b = (const float*)d_in[8];
  const float* w2b = (const float*)d_in[9];
  const float* b2b = (const float*)d_in[10];
  const float* wc  = (const float*)d_in[11];
  const float* bc  = (const float*)d_in[12];
  (void)n_in; (void)out_size; (void)ws_size;

  const int nNodes = in_sizes[0] / 3;
  const int nEdges = in_sizes[1] / 2;
  const int* esrc = eidx;
  const int* edst = eidx + nEdges;

  // Workspace carve-up (all L2-resident on MI455X: ~51 MB total)
  char* ws = (char*)d_ws;
  size_t off = 0;
  auto carve = [&](size_t bytes) {
    void* p = ws + off;
    off += (bytes + 255) & ~(size_t)255;
    return p;
  };
  const size_t nodeBytes = (size_t)nNodes * NCH * sizeof(float);
  float*  U  = (float*)carve(nodeBytes);
  float*  V  = (float*)carve(nodeBytes);
  float*  H1 = (float*)carve(nodeBytes);
  float*  H2 = (float*)carve(nodeBytes);
  float*  G  = (float*)carve(64 * 32 * sizeof(float));
  __bf16* wa = (__bf16*)carve(32 * 32 * sizeof(__bf16));
  __bf16* wb = (__bf16*)carve(32 * 32 * sizeof(__bf16));

  const int nodeThreads = nNodes * NCH;
  const int nodeBlocks = (nodeThreads + 255) / 256;
  const int edgeBlocks = 4096;  // 8 waves/block, grid-stride over 16-edge tiles

  wconv_kernel<<<8, 256, 0, stream>>>(w2a, w2b, wa, wb);

  // conv1
  pre1_kernel<<<nodeBlocks, 256, 0, stream>>>(pos, w1a, b1a, U, V, nNodes);
  zero_kernel<<<2048, 256, 0, stream>>>(H1, (long)nNodes * NCH);
  edge_conv_kernel<<<edgeBlocks, 256, 0, stream>>>(U, V, wa, b2a, esrc, edst,
                                                   H1, nEdges);
  // conv2
  pre2_kernel<<<nodeBlocks, 256, 0, stream>>>(pos, H1, w1b, b1b, U, V, nNodes);
  zero_kernel<<<2048, 256, 0, stream>>>(H2, (long)nNodes * NCH);
  edge_conv_kernel<<<edgeBlocks, 256, 0, stream>>>(U, V, wb, b2b, esrc, edst,
                                                   H2, nEdges);
  // pool + classify
  zero_kernel<<<8, 256, 0, stream>>>(G, 64 * 32);
  pool_kernel<<<256, 256, 0, stream>>>(H2, batch, G, nNodes);
  cls_kernel<<<(64 * 40 + 255) / 256, 256, 0, stream>>>(G, wc, bc,
                                                        (float*)d_out);
}
